// highwayNet_23106924052607
// MI455X (gfx1250) — matrix operand order repaired and checked by reading, not compiled, not run on hardware
//
#include <hip/hip_runtime.h>

typedef _Float16 h16;
typedef _Float16 v16h __attribute__((ext_vector_type(16)));
typedef _Float16 v8h  __attribute__((ext_vector_type(8)));
typedef float    v8f  __attribute__((ext_vector_type(8)));

#define DEV static __device__ __forceinline__

// ---------- constants ----------
#define BB      512
#define NN      39
#define NPIX    1521            // 39*39
#define NNODE   19968           // B*39
#define E0      159744          // B*312
#define ET      179712          // E0 + NNODE (self loops)
#define KP1     96              // 78 padded to mult of 32
#define KP2     512
#define HC1     512             // 8 heads * 64

DEV float fixv(float x) {
    if (x != x) return 0.0f;
    if (x ==  __builtin_inff()) return  3.4028234663852886e38f;
    if (x == -__builtin_inff()) return -3.4028234663852886e38f;
    return x;
}
// order-preserving float<->uint for atomicMax-based segment max
DEV unsigned f2o(float f) { unsigned u = __float_as_uint(f); return (u & 0x80000000u) ? ~u : (u | 0x80000000u); }
DEV float    o2f(unsigned u) { return (u & 0x80000000u) ? __uint_as_float(u & 0x7fffffffu) : __uint_as_float(~u); }

// ---------- utility fills ----------
__global__ void k_fill_f32(float* p, float v, int n) {
    int i = blockIdx.x * blockDim.x + threadIdx.x;
    if (i < n) p[i] = v;
}
__global__ void k_fill_enc(unsigned* p, float v, int n) {
    int i = blockIdx.x * blockDim.x + threadIdx.x;
    if (i < n) p[i] = f2o(v);
}

// ---------- conv1 (1x1, 3->8) + BN + ReLU ----------
__global__ void k_conv1(const float* __restrict__ man, const float* __restrict__ ac,
                        const float* __restrict__ ve,
                        const float* __restrict__ w, const float* __restrict__ bias,
                        const float* __restrict__ g, const float* __restrict__ bb,
                        float* __restrict__ X8) {
    int i = blockIdx.x * blockDim.x + threadIdx.x;
    if (i >= BB * NPIX) return;
    int b = i / NPIX, p = i % NPIX;
    float x0 = fixv(man[i]), x1 = fixv(ac[i]), x2 = fixv(ve[i]);
    const float inv = rsqrtf(1.0f + 1e-5f);
    #pragma unroll
    for (int co = 0; co < 8; ++co) {
        float a = w[co * 3 + 0] * x0 + w[co * 3 + 1] * x1 + w[co * 3 + 2] * x2 + bias[co];
        a = a * (g[co] * inv) + bb[co];
        X8[((size_t)b * 8 + co) * NPIX + p] = a > 0.0f ? a : 0.0f;
    }
}

// ---------- conv2 (3x3 pad1, 8->16) + BN ----------
__global__ void k_conv2(const float* __restrict__ X8, const float* __restrict__ w,
                        const float* __restrict__ bias, const float* __restrict__ g,
                        const float* __restrict__ bb, float* __restrict__ X16) {
    __shared__ float sw[1152];
    for (int t = threadIdx.x; t < 1152; t += blockDim.x) sw[t] = w[t];
    __syncthreads();
    int i = blockIdx.x * blockDim.x + threadIdx.x;
    if (i >= BB * 16 * NPIX) return;
    int wp = i % NN;
    int hp = (i / NN) % NN;
    int co = (i / NPIX) % 16;
    int b  = i / (16 * NPIX);
    float acc = 0.0f;
    for (int ci = 0; ci < 8; ++ci) {
        const float* base = X8 + ((size_t)b * 8 + ci) * NPIX;
        #pragma unroll
        for (int dy = 0; dy < 3; ++dy) {
            int hh = hp + dy - 1;
            if (hh < 0 || hh >= NN) continue;
            #pragma unroll
            for (int dx = 0; dx < 3; ++dx) {
                int ww = wp + dx - 1;
                if (ww < 0 || ww >= NN) continue;
                acc += sw[((co * 8 + ci) * 3 + dy) * 3 + dx] * base[hh * NN + ww];
            }
        }
    }
    X16[i] = (acc + bias[co]) * (g[co] * rsqrtf(1.0f + 1e-5f)) + bb[co];
}

// ---------- GRU over rows, hidden=1, one thread per (b, column) ----------
__global__ void k_gru(const float* __restrict__ X16, const float* __restrict__ wih,
                      const float* __restrict__ whh, const float* __restrict__ bih,
                      const float* __restrict__ bhh, float* __restrict__ Y) {
    __shared__ float swih[48], swhh[3], sbih[3], sbhh[3];
    if (threadIdx.x < 48) swih[threadIdx.x] = wih[threadIdx.x];
    if (threadIdx.x < 3) {
        swhh[threadIdx.x] = whh[threadIdx.x];
        sbih[threadIdx.x] = bih[threadIdx.x];
        sbhh[threadIdx.x] = bhh[threadIdx.x];
    }
    __syncthreads();
    int i = blockIdx.x * blockDim.x + threadIdx.x;
    if (i >= NNODE) return;
    int b = i / NN, w = i % NN;
    const float* xb = X16 + (size_t)b * 16 * NPIX;
    float h = 0.0f;
    for (int t = 0; t < NN; ++t) {
        float gi0 = sbih[0], gi1 = sbih[1], gi2 = sbih[2];
        #pragma unroll
        for (int c = 0; c < 16; ++c) {
            float x = xb[c * NPIX + t * NN + w];
            gi0 += swih[c] * x;
            gi1 += swih[16 + c] * x;
            gi2 += swih[32 + c] * x;
        }
        float gh0 = h * swhh[0] + sbhh[0];
        float gh1 = h * swhh[1] + sbhh[1];
        float gh2 = h * swhh[2] + sbhh[2];
        float r = 1.0f / (1.0f + expf(-(gi0 + gh0)));
        float z = 1.0f / (1.0f + expf(-(gi1 + gh1)));
        float n = tanhf(gi2 + r * gh2);
        h = (1.0f - z) * n + z * h;
        Y[(size_t)i * NN + t] = h;
    }
}

// ---------- build node features (78 padded to 96) directly as f16 ----------
__global__ void k_nodes(const float* __restrict__ man, const float* __restrict__ mask,
                        const float* __restrict__ Y, h16* __restrict__ AH) {
    int i = blockIdx.x * blockDim.x + threadIdx.x;
    if (i >= NNODE * KP1) return;
    int node = i / KP1, c = i % KP1;
    int b = node / NN, j = node % NN;
    float m = mask[b * NN + j];
    float v = 0.0f;
    if (c < NN)      v = fixv(man[((size_t)b * NN + c) * NN + j]) * m;
    else if (c < 78) v = Y[(size_t)node * NN + (c - NN)] * m;
    AH[i] = (h16)v;
}

// ---------- pack weights transposed+padded to f16 (Wt[N][Kp]) ----------
__global__ void k_packw1(const float* __restrict__ wl, const float* __restrict__ wr,
                         h16* __restrict__ tl, h16* __restrict__ tr) {
    int i = blockIdx.x * blockDim.x + threadIdx.x;
    if (i >= 512 * KP1) return;
    int n = i / KP1, k = i % KP1;
    tl[i] = (h16)((k < 78) ? wl[k * 512 + n] : 0.0f);
    tr[i] = (h16)((k < 78) ? wr[k * 512 + n] : 0.0f);
}
__global__ void k_packw2(const float* __restrict__ wl, const float* __restrict__ wr,
                         h16* __restrict__ tl, h16* __restrict__ tr) {
    int i = blockIdx.x * blockDim.x + threadIdx.x;
    if (i >= 64 * KP2) return;
    int n = i / KP2, k = i % KP2;
    tl[i] = (h16)wl[k * 64 + n];
    tr[i] = (h16)wr[k * 64 + n];
}

// ---------- load one 16x32 f16 A fragment (CDNA5 WMMA A layout) ----------
DEV v16h load_afrag(const h16* arow, int kk) {
    v8h lo = *(const v8h*)(arow + kk);        // K = kk + 8*half .. +7
    v8h hi = *(const v8h*)(arow + kk + 16);   // K = kk + 16 + 8*half .. +7
    v16h a;
    #pragma unroll
    for (int q = 0; q < 8; ++q) { a[q] = lo[q]; a[q + 8] = hi[q]; }
    return a;
}

// ---------- WMMA GEMM: C[M,N] = A[M,KP](f16) x Wt[N,KP]^T(f16) ----------
// One wave computes a 32x64 tile: 2 A-frags x 4 B-frags -> 8 WMMAs per K-step,
// 8x register-level reuse vs. the naive 16x16-per-wave version.
template <int KP>
__global__ __launch_bounds__(256) void k_gemm_wmma(const h16* __restrict__ A,
                                                   const h16* __restrict__ Wt,
                                                   float* __restrict__ C,
                                                   int Mt, int Nt) {   // Mt = M/32, Nt = N/64
    int wave = threadIdx.x >> 5;
    int lane = threadIdx.x & 31;
    int tile = blockIdx.x * 8 + wave;
    if (tile >= Mt * Nt) return;               // grids divide evenly -> wave-uniform
    int tm = tile / Nt, tn = tile % Nt;
    int half = lane >> 4, mr = lane & 15;
    int N = Nt * 64;

    const h16* arow0 = A + (size_t)(tm * 32 + mr) * KP + 8 * half;
    const h16* arow1 = arow0 + (size_t)16 * KP;
    const h16* brow  = Wt + (size_t)(tn * 64 + mr) * KP + 8 * half;

    v8f c[2][4];
    #pragma unroll
    for (int i = 0; i < 2; ++i)
        #pragma unroll
        for (int j = 0; j < 4; ++j) c[i][j] = (v8f){};

    for (int kk = 0; kk < KP; kk += 32) {
        __builtin_prefetch(arow0 + kk + 64, 0, 0);
        __builtin_prefetch(arow1 + kk + 64, 0, 0);
        v16h a0 = load_afrag(arow0, kk);
        v16h a1 = load_afrag(arow1, kk);
        #pragma unroll
        for (int j = 0; j < 4; ++j) {
            v16h bf = load_afrag(brow + (size_t)j * 16 * KP, kk);
            c[0][j] = __builtin_amdgcn_wmma_f32_16x16x32_f16(false, a0, false, bf,
                                                             (short)0, c[0][j], false, false);
            c[1][j] = __builtin_amdgcn_wmma_f32_16x16x32_f16(false, a1, false, bf,
                                                             (short)0, c[1][j], false, false);
        }
    }
    // C/D layout: row = r + 8*half (+16*i), col = mr (+16*j)
    #pragma unroll
    for (int i = 0; i < 2; ++i) {
        float* crow = C + (size_t)(tm * 32 + i * 16 + 8 * half) * N + tn * 64 + mr;
        #pragma unroll
        for (int j = 0; j < 4; ++j)
            #pragma unroll
            for (int r = 0; r < 8; ++r)
                crow[(size_t)r * N + j * 16] = c[i][j][r];
    }
}

// ---------- GATv2 edge logits + segment max ----------
__global__ void k_logit(const int* __restrict__ ei, const float* __restrict__ xl,
                        const float* __restrict__ xr, const float* __restrict__ att,
                        float* __restrict__ elog, unsigned* __restrict__ mmax,
                        int H, int C) {
    int i = blockIdx.x * blockDim.x + threadIdx.x;
    if (i >= ET * H) return;
    int e = i / H, h = i % H;
    int s, d;
    if (e < E0) { s = ei[e]; d = ei[E0 + e]; } else { s = d = e - E0; }
    int HC = H * C;
    const float* pl = xl + (size_t)s * HC + h * C;
    const float* pr = xr + (size_t)d * HC + h * C;
    const float* pa = att + h * C;
    float acc = 0.0f;
    for (int c = 0; c < C; ++c) {
        float t = pl[c] + pr[c];
        t = t > 0.0f ? t : 0.2f * t;            // LeakyReLU(0.2)
        acc += t * pa[c];
    }
    elog[i] = acc;
    atomicMax(mmax + (size_t)d * H + h, f2o(acc));
}

// ---------- exp(logit - max) + segment sum ----------
__global__ void k_expnorm(const int* __restrict__ ei, float* __restrict__ elog,
                          const unsigned* __restrict__ mmax, float* __restrict__ ssum,
                          int H) {
    int i = blockIdx.x * blockDim.x + threadIdx.x;
    if (i >= ET * H) return;
    int e = i / H, h = i % H;
    int d = (e < E0) ? ei[E0 + e] : (e - E0);
    float m = o2f(mmax[(size_t)d * H + h]);
    if (!(m > -1e37f && m < 1e37f)) m = 0.0f;   // jnp.where(isfinite(m), m, 0)
    float p = expf(elog[i] - m);
    elog[i] = p;
    atomicAdd(ssum + (size_t)d * H + h, p);
}

// ---------- weighted scatter-add aggregation ----------
__global__ void k_agg(const int* __restrict__ ei, const float* __restrict__ elog,
                      const float* __restrict__ xl, float* __restrict__ out,
                      int H, int C) {
    long i = (long)blockIdx.x * blockDim.x + threadIdx.x;
    long tot = (long)ET * H * C;
    if (i >= tot) return;
    int c = (int)(i % C);
    int h = (int)((i / C) % H);
    int e = (int)(i / ((long)H * C));
    int s, d;
    if (e < E0) { s = ei[e]; d = ei[E0 + e]; } else { s = d = e - E0; }
    int HC = H * C;
    float v = elog[(size_t)e * H + h] * xl[(size_t)s * HC + h * C + c];
    atomicAdd(out + (size_t)d * HC + h * C + c, v);
}

// ---------- finalize layer 1: /sum + bias, ELU, pack to f16 ----------
__global__ void k_final1(const float* __restrict__ out, const float* __restrict__ ssum,
                         const float* __restrict__ bias, h16* __restrict__ h1h) {
    int i = blockIdx.x * blockDim.x + threadIdx.x;
    if (i >= NNODE * HC1) return;
    int n = i >> 9, hc = i & 511, h = hc >> 6;
    float v = out[i] / (ssum[n * 8 + h] + 1e-16f) + bias[hc];
    v = v > 0.0f ? v : (expf(v) - 1.0f);        // ELU
    h1h[i] = (h16)v;
}

// ---------- finalize layer 2 (in-place on d_out) ----------
__global__ void k_final2(float* __restrict__ out, const float* __restrict__ ssum,
                         const float* __restrict__ bias) {
    int i = blockIdx.x * blockDim.x + threadIdx.x;
    if (i >= NNODE * 64) return;
    int n = i >> 6, c = i & 63;
    out[i] = out[i] / (ssum[n] + 1e-16f) + bias[c];
}

extern "C" void kernel_launch(void* const* d_in, const int* in_sizes, int n_in,
                              void* d_out, int out_size, void* d_ws, size_t ws_size,
                              hipStream_t stream) {
    const int*   ei    = (const int*)d_in[0];
    const float* ve    = (const float*)d_in[1];
    const float* ac    = (const float*)d_in[2];
    const float* man   = (const float*)d_in[3];
    const float* mask  = (const float*)d_in[4];
    const float* c1w   = (const float*)d_in[5];
    const float* c1b   = (const float*)d_in[6];
    const float* bn1g  = (const float*)d_in[7];
    const float* bn1b  = (const float*)d_in[8];
    const float* c2w   = (const float*)d_in[9];
    const float* c2b   = (const float*)d_in[10];
    const float* bn2g  = (const float*)d_in[11];
    const float* bn2b  = (const float*)d_in[12];
    const float* gwih  = (const float*)d_in[13];
    const float* gwhh  = (const float*)d_in[14];
    const float* gbih  = (const float*)d_in[15];
    const float* gbhh  = (const float*)d_in[16];
    const float* g1wl  = (const float*)d_in[17];
    const float* g1wr  = (const float*)d_in[18];
    const float* g1att = (const float*)d_in[19];
    const float* g1bias= (const float*)d_in[20];
    const float* g2wl  = (const float*)d_in[21];
    const float* g2wr  = (const float*)d_in[22];
    const float* g2att = (const float*)d_in[23];
    const float* g2bias= (const float*)d_in[24];
    float* outp = (float*)d_out;
    float* ws = (float*)d_ws;

    // workspace layout (float offsets); conv scratch region is reused for GEMM outputs
    float*    X8   = ws;                               //  6,230,016
    float*    X16  = ws + 6230016;                     // 12,460,032
    float*    XL1  = ws;                               // reuse X8+X16 region (10,223,616)
    float*    Y    = ws + 18690048;                    //    778,752
    h16*      AH   = (h16*)(ws + 19468800);            //  1,916,928 halves
    h16*      WT1L = (h16*)(ws + 20427264);            //     49,152 halves
    h16*      WT1R = (h16*)(ws + 20451840);
    h16*      WT2L = (h16*)(ws + 20476416);            //     32,768 halves
    h16*      WT2R = (h16*)(ws + 20492800);
    float*    XR1  = ws + 20509184;                    // 10,223,616
    h16*      H1H  = (h16*)XR1;                        // reuse after logit1
    float*    ELOG = ws + 30732800;                    //  1,437,696
    unsigned* MBUF = (unsigned*)(ws + 32170496);       //    159,744
    float*    SBUF = ws + 32330240;                    //    159,744
    float*    OUT1 = ws + 32489984;                    // 10,223,616 (end 42,713,600 floats)
    float*    XL2  = ws;                               // reuse after agg1
    float*    XR2  = ws + 1277952;

    // stage 1: convs
    k_conv1<<<(BB * NPIX + 255) / 256, 256, 0, stream>>>(man, ac, ve, c1w, c1b, bn1g, bn1b, X8);
    k_conv2<<<(BB * 16 * NPIX + 255) / 256, 256, 0, stream>>>(X8, c2w, c2b, bn2g, bn2b, X16);
    // stage 2: GRU
    k_gru<<<NNODE / 256, 256, 0, stream>>>(X16, gwih, gwhh, gbih, gbhh, Y);
    // stage 3: node features + weight packing
    k_nodes<<<(NNODE * KP1) / 256, 256, 0, stream>>>(man, mask, Y, AH);
    k_packw1<<<(512 * KP1 + 255) / 256, 256, 0, stream>>>(g1wl, g1wr, WT1L, WT1R);
    k_packw2<<<(64 * KP2 + 255) / 256, 256, 0, stream>>>(g2wl, g2wr, WT2L, WT2R);
    // stage 4: GAT layer 1 transforms via WMMA (624 x 8 tiles of 32x64 each)
    k_gemm_wmma<KP1><<<624, 256, 0, stream>>>(AH, WT1L, XL1, 624, 8);
    k_gemm_wmma<KP1><<<624, 256, 0, stream>>>(AH, WT1R, XR1, 624, 8);
    // stage 5: GAT layer 1 attention softmax + aggregation
    k_fill_enc<<<624, 256, 0, stream>>>(MBUF, -3.0e38f, NNODE * 8);
    k_fill_f32<<<624, 256, 0, stream>>>(SBUF, 0.0f, NNODE * 8);
    k_fill_f32<<<39936, 256, 0, stream>>>(OUT1, 0.0f, NNODE * HC1);
    k_logit<<<(ET * 8) / 256, 256, 0, stream>>>(ei, XL1, XR1, g1att, ELOG, MBUF, 8, 64);
    k_expnorm<<<(ET * 8) / 256, 256, 0, stream>>>(ei, ELOG, MBUF, SBUF, 8);
    k_agg<<<((long)ET * 8 * 64) / 256, 256, 0, stream>>>(ei, ELOG, XL1, OUT1, 8, 64);
    k_final1<<<(NNODE * HC1) / 256, 256, 0, stream>>>(OUT1, SBUF, g1bias, H1H);
    // stage 6: GAT layer 2 transforms via WMMA (624 x 1 tiles of 32x64 each)
    k_gemm_wmma<KP2><<<78, 256, 0, stream>>>(H1H, WT2L, XL2, 624, 1);
    k_gemm_wmma<KP2><<<78, 256, 0, stream>>>(H1H, WT2R, XR2, 624, 1);
    // stage 7: GAT layer 2 attention + aggregation into d_out
    k_fill_enc<<<78, 256, 0, stream>>>(MBUF, -3.0e38f, NNODE);
    k_fill_f32<<<78, 256, 0, stream>>>(SBUF, 0.0f, NNODE);
    k_fill_f32<<<(NNODE * 64) / 256, 256, 0, stream>>>(outp, 0.0f, NNODE * 64);
    k_logit<<<ET / 256, 256, 0, stream>>>(ei, XL2, XR2, g2att, ELOG, MBUF, 1, 64);
    k_expnorm<<<ET / 256, 256, 0, stream>>>(ei, ELOG, MBUF, SBUF, 1);
    k_agg<<<((long)ET * 64) / 256, 256, 0, stream>>>(ei, ELOG, XL2, outp, 1, 64);
    k_final2<<<(NNODE * 64) / 256, 256, 0, stream>>>(outp, SBUF, g2bias);
    (void)in_sizes; (void)n_in; (void)out_size; (void)ws_size;
}
